// TransformerEncoderBlock_50818053046417
// MI455X (gfx1250) — matrix operand order repaired and checked by reading, not compiled, not run on hardware
//
#include <hip/hip_runtime.h>
#include <hip/hip_bf16.h>
#include <math.h>

// ---------------------------------------------------------------------------
// Transformer encoder block for MI455X (gfx1250, wave32, WMMA).
// GEMMs: V_WMMA_F32_16X16X32_F16 with double-buffered LDS fed by
// GLOBAL_LOAD_ASYNC_TO_LDS_B128 (ASYNCcnt pipeline).
// Attention: flash-style, WMMA for both QK^T and PV, ds_swizzle reductions.
// ---------------------------------------------------------------------------

typedef _Float16 half_t;
typedef __attribute__((ext_vector_type(16))) _Float16 v16h;
typedef __attribute__((ext_vector_type(8)))  _Float16 v8h;
typedef __attribute__((ext_vector_type(4)))  _Float16 v4h;
typedef __attribute__((ext_vector_type(8)))  float    v8f;

#define EMBED 1024
#define HEADS 16
#define HDIM  64
#define MLPH  4096
#define BB    2
#define SS    2048
#define MM    (BB * SS)   // 4096 rows

__device__ __forceinline__ int lane_id() { return threadIdx.x & 31; }

__device__ __forceinline__ v8f zero_v8f() {
  v8f z;
#pragma unroll
  for (int i = 0; i < 8; ++i) z[i] = 0.0f;
  return z;
}

// A fragment (16x32 f16): lanes 0-15 -> M=lane, K chunks {0..7, 16..23};
// lanes 16-31 -> M=lane-16, K chunks {8..15, 24..31}. Two 16B loads.
__device__ __forceinline__ v16h load_frag_a(const half_t* p, int ld) {
  const int lane = lane_id();
  const half_t* r = p + (size_t)(lane & 15) * ld + ((lane & 16) ? 8 : 0);
  union { v16h v; v8h h[2]; } u;
  u.h[0] = *(const v8h*)(r);
  u.h[1] = *(const v8h*)(r + 16);
  return u.v;
}

// B fragment (32x16 f16) from a [N][K] (K-contiguous) buffer:
// lanes 0-15 -> N=lane, K chunks {0..7, 16..23}; lanes 16-31 -> N=lane-16, K chunks {8..15, 24..31}.
__device__ __forceinline__ v16h load_frag_b(const half_t* p, int ld) {
  const int lane = lane_id();
  const half_t* r = p + (size_t)(lane & 15) * ld + ((lane & 16) ? 8 : 0);
  union { v16h v; v8h h[2]; } u;
  u.h[0] = *(const v8h*)(r);
  u.h[1] = *(const v8h*)(r + 16);
  return u.v;
}

__device__ __forceinline__ v8f wmma_f16(v16h a, v16h b, v8f c) {
  // (neg_a, A, neg_b, B, c_mod, C, reuse_a, reuse_b)
  return __builtin_amdgcn_wmma_f32_16x16x32_f16(false, a, false, b, (short)0, c,
                                                false, false);
}

// XOR lane shuffle via ds_swizzle (immediate mask, no index VGPRs).
// offset[14:10]=xor, [9:5]=or, [4:0]=and; and=0x1f keeps full 32-lane group.
template <int XM>
__device__ __forceinline__ float swz_xor(float v) {
  return __int_as_float(
      __builtin_amdgcn_ds_swizzle(__float_as_int(v), (XM << 10) | 0x1f));
}

// ---- CDNA5 async global->LDS copy (ASYNCcnt) ------------------------------
__device__ __forceinline__ void async_copy_b128(unsigned lds_byte_off,
                                                const half_t* g) {
  unsigned long long ga = (unsigned long long)(uintptr_t)g;
  asm volatile("global_load_async_to_lds_b128 %0, %1, off"
               :: "v"(lds_byte_off), "v"(ga) : "memory");
}
__device__ __forceinline__ void wait_async0() {
  asm volatile("s_wait_asynccnt 0" ::: "memory");
}

// ---------------------------------------------------------------------------
// Weight pre-transpose + f32->f16 cast:  src[K][N] f32  ->  dst[N][K] f16
// ---------------------------------------------------------------------------
__global__ __launch_bounds__(256)
void transpose_cast_kernel(const float* __restrict__ src, half_t* __restrict__ dst,
                           int K, int N) {
  __shared__ float tile[32][33];
  const int bn = blockIdx.x * 32, bk = blockIdx.y * 32;
  const int tx = threadIdx.x & 31, ty = threadIdx.x >> 5;
#pragma unroll
  for (int i = 0; i < 32; i += 8)
    tile[ty + i][tx] = src[(size_t)(bk + ty + i) * N + bn + tx];
  __syncthreads();
#pragma unroll
  for (int i = 0; i < 32; i += 8)
    dst[(size_t)(bn + ty + i) * K + bk + tx] = (half_t)tile[tx][ty + i];
}

// ---------------------------------------------------------------------------
// Fused LayerNorm (f32 in) -> f16 out. One 256-thread block per row (D=1024).
// ---------------------------------------------------------------------------
__global__ __launch_bounds__(256)
void layernorm_cast_kernel(const float* __restrict__ x, const float* __restrict__ g,
                           const float* __restrict__ be, half_t* __restrict__ out) {
  const int row = blockIdx.x;
  const int tid = threadIdx.x;
  const float4 v = ((const float4*)(x + (size_t)row * EMBED))[tid];
  float s  = v.x + v.y + v.z + v.w;
  float sq = v.x * v.x + v.y * v.y + v.z * v.z + v.w * v.w;
#pragma unroll
  for (int off = 16; off > 0; off >>= 1) {
    s  += __shfl_xor(s, off, 32);
    sq += __shfl_xor(sq, off, 32);
  }
  __shared__ float rs[8], rq[8];
  if ((tid & 31) == 0) { rs[tid >> 5] = s; rq[tid >> 5] = sq; }
  __syncthreads();
  s = 0.0f; sq = 0.0f;
#pragma unroll
  for (int w = 0; w < 8; ++w) { s += rs[w]; sq += rq[w]; }
  const float mu   = s * (1.0f / EMBED);
  const float var  = sq * (1.0f / EMBED) - mu * mu;
  const float rstd = rsqrtf(var + 1e-5f);
  const int c0 = tid * 4;
  v4h o;
  o[0] = (half_t)((v.x - mu) * rstd * g[c0 + 0] + be[c0 + 0]);
  o[1] = (half_t)((v.y - mu) * rstd * g[c0 + 1] + be[c0 + 1]);
  o[2] = (half_t)((v.z - mu) * rstd * g[c0 + 2] + be[c0 + 2]);
  o[3] = (half_t)((v.w - mu) * rstd * g[c0 + 3] + be[c0 + 3]);
  ((v4h*)(out + (size_t)row * EMBED))[tid] = o;
}

// ---------------------------------------------------------------------------
// Tiled WMMA GEMM:  C[M][N](f32 acc) = A[M][K](f16,row) x BT[N][K](f16)
// Block tile 128x128x32, 8 waves (2Mx4N), wave tile 64x32.
// Double-buffered LDS (40-half padded rows: 16B-aligned, conflict-free),
// filled by GLOBAL_LOAD_ASYNC_TO_LDS_B128; compute overlaps next-tile DMA.
// ---------------------------------------------------------------------------
#define BM 128
#define BN 128
#define BKK 32
#define LDST 40
#define TILE_BYTES (BM * LDST * 2)

enum { EPI_BIAS_RES_F32 = 0, EPI_BIAS_GELU_F16 = 1, EPI_QKV_SCATTER = 2 };

template <int EPI>
__global__ __launch_bounds__(256)
void gemm_f16_kernel(const half_t* __restrict__ A, const half_t* __restrict__ BT,
                     int M, int N, int K,
                     const float* __restrict__ bias, const float* __restrict__ res,
                     float* __restrict__ outF32, half_t* __restrict__ outF16,
                     half_t* __restrict__ outQ, half_t* __restrict__ outK,
                     half_t* __restrict__ outVT) {
  __shared__ half_t sA[2][BM * LDST];
  __shared__ half_t sB[2][BN * LDST];
  const int tid  = threadIdx.x;
  const int lane = tid & 31, wave = tid >> 5;
  const int wm = wave & 1, wn = wave >> 1;            // 2 x 4 wave grid
  const int n0 = blockIdx.x * BN, m0 = blockIdx.y * BM;
  const int lr = tid >> 1;                            // 0..127 tile row
  const int lc = (tid & 1) * 16;                      // 0 or 16 (halves)

  v8f c[4][2];
#pragma unroll
  for (int i = 0; i < 4; ++i)
#pragma unroll
    for (int j = 0; j < 2; ++j) c[i][j] = zero_v8f();

  const half_t* gA = A  + (size_t)(m0 + lr) * K + lc;
  const half_t* gB = BT + (size_t)(n0 + lr) * K + lc;
  // LDS byte offsets (generic-pointer low 32 bits == LDS offset on CDNA5)
  const unsigned aoff0 = (unsigned)(uintptr_t)&sA[0][lr * LDST + lc];
  const unsigned boff0 = (unsigned)(uintptr_t)&sB[0][lr * LDST + lc];

  // prologue: async-fill buffer 0 with tile 0
  async_copy_b128(aoff0,      gA);
  async_copy_b128(aoff0 + 16, gA + 8);
  async_copy_b128(boff0,      gB);
  async_copy_b128(boff0 + 16, gB + 8);

  const int NIT = K / BKK;
  for (int it = 0; it < NIT; ++it) {
    wait_async0();          // my async fills of the buffer I'm about to read
    __syncthreads();        // everyone's fills done; prev-iter readers done
    if (it + 1 < NIT) {     // overlap next-tile DMA with this tile's WMMAs
      const int nb = (it + 1) & 1;
      const half_t* ga = gA + (it + 1) * BKK;
      const half_t* gb = gB + (it + 1) * BKK;
      async_copy_b128(aoff0 + nb * TILE_BYTES,      ga);
      async_copy_b128(aoff0 + nb * TILE_BYTES + 16, ga + 8);
      async_copy_b128(boff0 + nb * TILE_BYTES,      gb);
      async_copy_b128(boff0 + nb * TILE_BYTES + 16, gb + 8);
      if (it + 2 < NIT) {   // L2 prefetch two tiles ahead
        __builtin_prefetch(gA + (it + 2) * BKK, 0, 1);
        __builtin_prefetch(gB + (it + 2) * BKK, 0, 1);
      }
    }
    const int cur = it & 1;
    v16h af[4], bf[2];
#pragma unroll
    for (int i = 0; i < 4; ++i)
      af[i] = load_frag_a(&sA[cur][(wm * 64 + i * 16) * LDST], LDST);
#pragma unroll
    for (int j = 0; j < 2; ++j)
      bf[j] = load_frag_b(&sB[cur][(wn * 32 + j * 16) * LDST], LDST);
#pragma unroll
    for (int i = 0; i < 4; ++i)
#pragma unroll
      for (int j = 0; j < 2; ++j) c[i][j] = wmma_f16(af[i], bf[j], c[i][j]);
  }

  // ---- epilogue (C layout: N = lane&15; M = vgpr + (lane&16 ? 8 : 0)) ----
  const int nl = lane & 15;
  const int r0 = (lane & 16) ? 8 : 0;
#pragma unroll
  for (int i = 0; i < 4; ++i) {
#pragma unroll
    for (int j = 0; j < 2; ++j) {
      const int fm = m0 + wm * 64 + i * 16 + r0;
      const int fn = n0 + wn * 32 + j * 16 + nl;
#pragma unroll
      for (int r = 0; r < 8; ++r) {
        const int gm = fm + r;
        float v = c[i][j][r];
        if (EPI == EPI_BIAS_RES_F32) {
          v += bias[fn] + res[(size_t)gm * N + fn];
          outF32[(size_t)gm * N + fn] = v;
        } else if (EPI == EPI_BIAS_GELU_F16) {
          v += bias[fn];
          v = 0.5f * v * (1.0f + erff(v * 0.70710678118654752f));   // exact GELU
          outF16[(size_t)gm * N + fn] = (half_t)v;
        } else {  // EPI_QKV_SCATTER: split [*,3072] into q / k / v^T layouts
          const int which = fn >> 10;          // 0=q 1=k 2=v
          const int cc = fn & 1023;
          const int hh = cc >> 6;
          const int dh = cc & 63;
          const int b  = gm >> 11;             // S = 2048
          const int s  = gm & 2047;
          const size_t bh = (size_t)b * HEADS + hh;
          if (which == 0)
            outQ[(bh * SS + s) * HDIM + dh] = (half_t)(v * 0.125f);  // 1/sqrt(64)
          else if (which == 1)
            outK[(bh * SS + s) * HDIM + dh] = (half_t)v;
          else
            outVT[(bh * HDIM + dh) * SS + s] = (half_t)v;
        }
      }
    }
  }
}

// ---------------------------------------------------------------------------
// Flash attention: grid (S/128, B*H), 8 waves/block, each wave = 16 q rows.
// S = Q K^T and O += P V via WMMA; online softmax with batched ds_swizzle
// XOR reductions (8 in flight per wait instead of 2).
// ---------------------------------------------------------------------------
__global__ __launch_bounds__(256)
void flash_attn_kernel(const half_t* __restrict__ q, const half_t* __restrict__ k,
                       const half_t* __restrict__ vT, half_t* __restrict__ attnOut) {
  __shared__ half_t sP[8 * 16 * 64];          // 16KB, one 16x64 tile per wave
  const int lane = threadIdx.x & 31, wave = threadIdx.x >> 5;
  const int bh = blockIdx.y;
  const int b = bh >> 4, hh = bh & 15;
  const int qrow = blockIdx.x * 128 + wave * 16;

  const half_t* qp = q + ((size_t)bh * SS + qrow) * HDIM;
  const v16h qa0 = load_frag_a(qp, HDIM);       // Dh 0..31
  const v16h qa1 = load_frag_a(qp + 32, HDIM);  // Dh 32..63

  v8f acc[4];
  float mrow[8], lrow[8];
#pragma unroll
  for (int jd = 0; jd < 4; ++jd) acc[jd] = zero_v8f();
#pragma unroll
  for (int r = 0; r < 8; ++r) { mrow[r] = -3.0e38f; lrow[r] = 0.0f; }

  half_t* myP = sP + wave * 16 * 64;
  const half_t* kbase = k  + (size_t)bh * SS * HDIM;
  const half_t* vbase = vT + (size_t)bh * HDIM * SS;
  const int nl = lane & 15;
  const int r0 = (lane & 16) ? 8 : 0;

  for (int kb = 0; kb < SS; kb += 64) {
    // ---- S = Q K^T for this 16q x 64k block ----
    v8f s[4];
#pragma unroll
    for (int j = 0; j < 4; ++j) {
      const half_t* kp = kbase + (size_t)(kb + j * 16) * HDIM;
      s[j] = wmma_f16(qa0, load_frag_b(kp, HDIM), zero_v8f());
      s[j] = wmma_f16(qa1, load_frag_b(kp + 32, HDIM), s[j]);
    }
    // ---- row max: batch all 8 rows per XOR pass ----
    float mx[8];
#pragma unroll
    for (int r = 0; r < 8; ++r)
      mx[r] = fmaxf(fmaxf(s[0][r], s[1][r]), fmaxf(s[2][r], s[3][r]));
#pragma unroll
    for (int r = 0; r < 8; ++r) mx[r] = fmaxf(mx[r], swz_xor<8>(mx[r]));
#pragma unroll
    for (int r = 0; r < 8; ++r) mx[r] = fmaxf(mx[r], swz_xor<4>(mx[r]));
#pragma unroll
    for (int r = 0; r < 8; ++r) mx[r] = fmaxf(mx[r], swz_xor<2>(mx[r]));
#pragma unroll
    for (int r = 0; r < 8; ++r) mx[r] = fmaxf(mx[r], swz_xor<1>(mx[r]));
    // ---- online softmax update ----
    float rsum[8];
#pragma unroll
    for (int r = 0; r < 8; ++r) {
      const float mnew = fmaxf(mrow[r], mx[r]);
      const float corr = __expf(mrow[r] - mnew);
      mrow[r] = mnew;
      float rs = 0.0f;
#pragma unroll
      for (int j = 0; j < 4; ++j) {
        const float p = __expf(s[j][r] - mnew);
        s[j][r] = p;
        rs += p;
      }
      rsum[r] = rs;
      lrow[r] *= corr;
#pragma unroll
      for (int jd = 0; jd < 4; ++jd) acc[jd][r] *= corr;
    }
#pragma unroll
    for (int r = 0; r < 8; ++r) rsum[r] += swz_xor<8>(rsum[r]);
#pragma unroll
    for (int r = 0; r < 8; ++r) rsum[r] += swz_xor<4>(rsum[r]);
#pragma unroll
    for (int r = 0; r < 8; ++r) rsum[r] += swz_xor<2>(rsum[r]);
#pragma unroll
    for (int r = 0; r < 8; ++r) rsum[r] += swz_xor<1>(rsum[r]);
#pragma unroll
    for (int r = 0; r < 8; ++r) lrow[r] += rsum[r];
    // ---- stage P to wave-private LDS, reload as A fragments ----
#pragma unroll
    for (int j = 0; j < 4; ++j)
#pragma unroll
      for (int r = 0; r < 8; ++r)
        myP[(r0 + r) * 64 + j * 16 + nl] = (half_t)s[j][r];
    asm volatile("s_wait_dscnt 0" ::: "memory");  // same-wave cross-lane LDS RAW
    const v16h pa0 = load_frag_a(myP, 64);        // keys 0..31 of block
    const v16h pa1 = load_frag_a(myP + 32, 64);   // keys 32..63
    // ---- O += P V   (B fragments from v^T: [Dh][S], key-contiguous) ----
#pragma unroll
    for (int jd = 0; jd < 4; ++jd) {
      const half_t* vp = vbase + (size_t)(jd * 16) * SS + kb;
      acc[jd] = wmma_f16(pa0, load_frag_b(vp, SS), acc[jd]);
      acc[jd] = wmma_f16(pa1, load_frag_b(vp + 32, SS), acc[jd]);
    }
  }

  // ---- normalize and write [B,S,H*Dh] f16 ----
#pragma unroll
  for (int jd = 0; jd < 4; ++jd)
#pragma unroll
    for (int r = 0; r < 8; ++r) {
      const int row = qrow + r0 + r;
      const float o = acc[jd][r] / lrow[r];
      attnOut[((size_t)(b * SS + row)) * EMBED + hh * HDIM + jd * 16 + nl] = (half_t)o;
    }
}

// ---------------------------------------------------------------------------
// Host orchestration
// ---------------------------------------------------------------------------
extern "C" void kernel_launch(void* const* d_in, const int* in_sizes, int n_in,
                              void* d_out, int out_size, void* d_ws, size_t ws_size,
                              hipStream_t stream) {
  (void)in_sizes; (void)n_in; (void)out_size; (void)ws_size;
  const float* x      = (const float*)d_in[0];
  const float* w_qkv  = (const float*)d_in[1];
  const float* w_proj = (const float*)d_in[2];
  const float* b_proj = (const float*)d_in[3];
  const float* w_fc1  = (const float*)d_in[4];
  const float* b_fc1  = (const float*)d_in[5];
  const float* w_fc2  = (const float*)d_in[6];
  const float* b_fc2  = (const float*)d_in[7];
  const float* g1     = (const float*)d_in[8];
  const float* be1    = (const float*)d_in[9];
  const float* g2     = (const float*)d_in[10];
  const float* be2    = (const float*)d_in[11];
  float* out = (float*)d_out;

  char* ws = (char*)d_ws;
  size_t off = 0;
  auto alloc = [&](size_t bytes) -> char* {
    char* p = ws + off;
    off += (bytes + 255) & ~(size_t)255;
    return p;
  };
  half_t* hln    = (half_t*)alloc((size_t)MM * EMBED * 2);        // LN output (reused for LN2)
  half_t* wqkvT  = (half_t*)alloc((size_t)3 * EMBED * EMBED * 2); // [3072][1024] f16
  half_t* wprojT = (half_t*)alloc((size_t)EMBED * EMBED * 2);     // [1024][1024]
  half_t* wfc1T  = (half_t*)alloc((size_t)MLPH * EMBED * 2);      // [4096][1024]
  half_t* wfc2T  = (half_t*)alloc((size_t)EMBED * MLPH * 2);      // [1024][4096]
  half_t* qbuf   = (half_t*)alloc((size_t)MM * EMBED * 2);        // [B,H,S,Dh] (pre-scaled)
  half_t* kbuf   = (half_t*)alloc((size_t)MM * EMBED * 2);        // [B,H,S,Dh]
  half_t* vTbuf  = (half_t*)alloc((size_t)MM * EMBED * 2);        // [B,H,Dh,S]
  half_t* attnO  = (half_t*)alloc((size_t)MM * EMBED * 2);        // [B,S,D] f16
  float*  x1     = (float*)alloc((size_t)MM * EMBED * 4);         // residual 1 (f32)
  half_t* mlp1   = (half_t*)alloc((size_t)MM * MLPH * 2);         // gelu(fc1) f16

  const dim3 blk(256);

  // Weight pre-transpose + cast (all [K][N] f32 -> [N][K] f16)
  transpose_cast_kernel<<<dim3(3 * EMBED / 32, EMBED / 32), blk, 0, stream>>>(w_qkv,  wqkvT,  EMBED, 3 * EMBED);
  transpose_cast_kernel<<<dim3(EMBED / 32,     EMBED / 32), blk, 0, stream>>>(w_proj, wprojT, EMBED, EMBED);
  transpose_cast_kernel<<<dim3(MLPH / 32,      EMBED / 32), blk, 0, stream>>>(w_fc1,  wfc1T,  EMBED, MLPH);
  transpose_cast_kernel<<<dim3(EMBED / 32,     MLPH  / 32), blk, 0, stream>>>(w_fc2,  wfc2T,  MLPH,  EMBED);

  // LN1 -> h (f16)
  layernorm_cast_kernel<<<dim3(MM), blk, 0, stream>>>(x, g1, be1, hln);

  // qkv = h @ Wqkv, scattered into q / k / v^T (q pre-scaled by 1/sqrt(Dh))
  gemm_f16_kernel<EPI_QKV_SCATTER><<<dim3(3 * EMBED / BN, MM / BM), blk, 0, stream>>>(
      hln, wqkvT, MM, 3 * EMBED, EMBED, nullptr, nullptr, nullptr, nullptr,
      qbuf, kbuf, vTbuf);

  // flash attention -> attnO [B,S,D] f16
  flash_attn_kernel<<<dim3(SS / 128, BB * HEADS), blk, 0, stream>>>(qbuf, kbuf, vTbuf, attnO);

  // x1 = x + attnO @ Wproj + b_proj
  gemm_f16_kernel<EPI_BIAS_RES_F32><<<dim3(EMBED / BN, MM / BM), blk, 0, stream>>>(
      attnO, wprojT, MM, EMBED, EMBED, b_proj, x, x1, nullptr,
      nullptr, nullptr, nullptr);

  // LN2 -> h2 (reuses hln buffer; prior consumer already done on this stream)
  layernorm_cast_kernel<<<dim3(MM), blk, 0, stream>>>(x1, g2, be2, hln);

  // mlp1 = gelu(h2 @ Wfc1 + b_fc1)  (f16)
  gemm_f16_kernel<EPI_BIAS_GELU_F16><<<dim3(MLPH / BN, MM / BM), blk, 0, stream>>>(
      hln, wfc1T, MM, MLPH, EMBED, b_fc1, nullptr, nullptr, mlp1,
      nullptr, nullptr, nullptr);

  // out = x1 + mlp1 @ Wfc2 + b_fc2  (f32)
  gemm_f16_kernel<EPI_BIAS_RES_F32><<<dim3(EMBED / BN, MM / BM), blk, 0, stream>>>(
      mlp1, wfc2T, MM, EMBED, MLPH, b_fc2, x1, out, nullptr,
      nullptr, nullptr, nullptr);
}
